// AdditiveAttention_63419487093403
// MI455X (gfx1250) — compile-verified, matrix operand order checked by reading, not run on hardware
//
#include <hip/hip_runtime.h>
#include <hip/hip_bf16.h>

// ---------------------------------------------------------------------------
// Fused additive attention for MI455X (gfx1250, wave32, WMMA).
//
// Roofline: x = 256MB must be read exactly ONCE (HBM @ 23.3 TB/s -> ~11us).
// Score GEMM (8.6 GFLOP) runs through v_wmma_f32_16x16x32_f16 so compute
// never becomes the bottleneck (fp32 VALU could not keep up at 33 FLOP/B).
// One block per batch row; online softmax so no second pass over x.
// tanh uses the CDNA5 hardware TANH trans op (co-executes with WMMA).
// ---------------------------------------------------------------------------

typedef __attribute__((ext_vector_type(16))) _Float16 v16h;
typedef __attribute__((ext_vector_type(8)))  float    v8f;

constexpr int Bsz = 256, Ssz = 2048, Dsz = 128, Usz = 64;
constexpr int CHUNK  = 128;           // seq positions per block iteration
constexpr int NCHUNK = Ssz / CHUNK;   // 16
constexpr int KC = Dsz / 32;          // 4 K-chunks of 32 for f16 WMMA
constexpr int NT = Usz / 16;          // 4 N-tiles of 16

__device__ __forceinline__ float fast_tanh(float v) {
#if __has_builtin(__builtin_amdgcn_tanhf)
    return __builtin_amdgcn_tanhf(v);     // v_tanh_f32 (1 trans op)
#else
    return tanhf(v);                      // fallback: libm expansion
#endif
}

__global__ __launch_bounds__(256)
void additive_attn_fused(const float* __restrict__ x,
                         const float* __restrict__ W1,
                         const float* __restrict__ W2,
                         const float* __restrict__ bias,
                         float* __restrict__ out_ctx,
                         float* __restrict__ out_alpha)
{
    __shared__ float  s_scores[Ssz];   // all scores for this batch row (8KB)
    __shared__ float4 s_red[256];      // ctx partial reduction

    const int tid  = threadIdx.x;
    const int lane = tid & 31;
    const int wave = tid >> 5;
    const int b    = blockIdx.x;

    const float* __restrict__ xb = x + (size_t)b * Ssz * Dsz;

    const int nlo  = lane & 15;        // M (A) / N (B,C) index within half-wave
    const int hi   = lane >> 4;        // 0: lanes 0-15, 1: lanes 16-31
    const int koff = hi ? 8 : 0;       // K offset per ISA 16-bit A/B layout

    // ---- one-time: W1 B-fragments (16 tiles, f16), bias & W2 per column ----
    v16h  w1f[KC][NT];
    float bv[NT], w2v[NT];
    #pragma unroll
    for (int nt = 0; nt < NT; ++nt) {
        const int n = nt * 16 + nlo;
        bv[nt]  = bias[n];
        w2v[nt] = W2[n];                       // W2 is [U,1]
        #pragma unroll
        for (int kc = 0; kc < KC; ++kc) {
            v16h f;
            #pragma unroll
            for (int j = 0; j < 16; ++j) {
                const int k = kc * 32 + koff + (j < 8 ? j : 16 + (j - 8));
                f[j] = (_Float16)W1[k * Usz + n];
            }
            w1f[kc][nt] = f;
        }
    }

    // ---- online-softmax running state (block-uniform by construction) ----
    float  m_run = -__builtin_inff();
    float  l_run = 0.f;
    float4 acc   = make_float4(0.f, 0.f, 0.f, 0.f);   // ctx partial, 4 d's

    for (int c = 0; c < NCHUNK; ++c) {
        // ============ phase 1: scores for this wave's 16-row strip ========
        const int srow = c * CHUNK + wave * 16;
        const float* __restrict__ xr = xb + (size_t)(srow + nlo) * Dsz;

        v16h af[KC];                       // A-fragments: 16x128 of x as f16
        #pragma unroll
        for (int kc = 0; kc < KC; ++kc) {
            const float* p = xr + kc * 32 + koff;
            const float4 f0 = *(const float4*)(p);
            const float4 f1 = *(const float4*)(p + 4);
            const float4 f2 = *(const float4*)(p + 16);
            const float4 f3 = *(const float4*)(p + 20);
            v16h a;
            a[0]=(_Float16)f0.x; a[1]=(_Float16)f0.y; a[2]=(_Float16)f0.z; a[3]=(_Float16)f0.w;
            a[4]=(_Float16)f1.x; a[5]=(_Float16)f1.y; a[6]=(_Float16)f1.z; a[7]=(_Float16)f1.w;
            a[8]=(_Float16)f2.x; a[9]=(_Float16)f2.y; a[10]=(_Float16)f2.z; a[11]=(_Float16)f2.w;
            a[12]=(_Float16)f3.x; a[13]=(_Float16)f3.y; a[14]=(_Float16)f3.z; a[15]=(_Float16)f3.w;
            af[kc] = a;
        }

        float partial[8];
        #pragma unroll
        for (int v = 0; v < 8; ++v) partial[v] = 0.f;

        #pragma unroll
        for (int nt = 0; nt < NT; ++nt) {
            v8f cacc;
            #pragma unroll
            for (int v = 0; v < 8; ++v) cacc[v] = bv[nt];   // + b (pre-tanh)
            #pragma unroll
            for (int kc = 0; kc < KC; ++kc) {
                cacc = __builtin_amdgcn_wmma_f32_16x16x32_f16(
                    false, af[kc], false, w1f[kc][nt],
                    (short)0, cacc, false, false);
            }
            #pragma unroll
            for (int v = 0; v < 8; ++v)
                partial[v] += fast_tanh(cacc[v]) * w2v[nt]; // dot with W2
        }
        // reduce over U across each 16-lane half (M = v + 8*hi)
        #pragma unroll
        for (int msk = 1; msk < 16; msk <<= 1) {
            #pragma unroll
            for (int v = 0; v < 8; ++v)
                partial[v] += __shfl_xor(partial[v], msk, 32);
        }
        if (nlo == 0) {                       // lanes 0 and 16
            #pragma unroll
            for (int v = 0; v < 8; ++v)
                s_scores[srow + hi * 8 + v] = partial[v];
        }
        __syncthreads();

        // ===== phase 2: online softmax update (wave-redundant, uniform) ====
        const int cbase = c * CHUNK;
        float sc[4], smax = -__builtin_inff();
        #pragma unroll
        for (int j = 0; j < 4; ++j) {
            sc[j] = s_scores[cbase + lane + 32 * j];
            smax  = fmaxf(smax, sc[j]);
        }
        #pragma unroll
        for (int msk = 1; msk < 32; msk <<= 1)
            smax = fmaxf(smax, __shfl_xor(smax, msk, 32));
        const float m_new = fmaxf(m_run, smax);
        float sume = 0.f;
        #pragma unroll
        for (int j = 0; j < 4; ++j)
            sume += __expf(sc[j] - m_new);
        #pragma unroll
        for (int msk = 1; msk < 32; msk <<= 1)
            sume += __shfl_xor(sume, msk, 32);
        const float r = __expf(m_run - m_new);   // first chunk: exp(-inf)=0
        l_run = l_run * r + sume;
        m_run = m_new;

        // rescale ctx accumulator, add this wave's 16 positions (L0-hot rows)
        acc.x *= r; acc.y *= r; acc.z *= r; acc.w *= r;
        #pragma unroll
        for (int j = 0; j < 16; ++j) {
            const int i = wave * 16 + j;
            const float w = __expf(s_scores[cbase + i] - m_new); // LDS bcast
            const float4 xv =
                *(const float4*)(xb + (size_t)(cbase + i) * Dsz + lane * 4);
            acc.x += w * xv.x; acc.y += w * xv.y;
            acc.z += w * xv.z; acc.w += w * xv.w;
        }
        // next chunk writes a disjoint s_scores region -> no barrier needed
    }

    // ---- ctx: reduce the 8 per-wave partials for each feature group ----
    s_red[tid] = acc;
    __syncthreads();
    const float inv_l = 1.f / l_run;
    if (tid < 32) {
        float4 t = s_red[tid];
        #pragma unroll
        for (int g = 1; g < 8; ++g) {
            const float4 o = s_red[g * 32 + tid];
            t.x += o.x; t.y += o.y; t.z += o.z; t.w += o.w;
        }
        const float4 o4 = make_float4(t.x * inv_l, t.y * inv_l,
                                      t.z * inv_l, t.w * inv_l);
        *(float4*)(out_ctx + (size_t)b * Dsz + tid * 4) = o4;
    }

    // ---- alpha from resident LDS scores (no re-read of x) ----
    for (int i = tid; i < Ssz; i += 256)
        out_alpha[(size_t)b * Ssz + i] = __expf(s_scores[i] - m_run) * inv_l;
}

extern "C" void kernel_launch(void* const* d_in, const int* in_sizes, int n_in,
                              void* d_out, int out_size, void* d_ws, size_t ws_size,
                              hipStream_t stream) {
    (void)in_sizes; (void)n_in; (void)out_size; (void)d_ws; (void)ws_size;
    const float* x    = (const float*)d_in[0];
    const float* W1   = (const float*)d_in[1];
    const float* W2   = (const float*)d_in[2];
    const float* bias = (const float*)d_in[3];
    float* out_ctx   = (float*)d_out;                         // [B, D]
    float* out_alpha = (float*)d_out + (size_t)Bsz * Dsz;     // [B, S, 1]
    additive_attn_fused<<<Bsz, 256, 0, stream>>>(x, W1, W2, bias,
                                                 out_ctx, out_alpha);
}
